// Procedure_60017872994694
// MI455X (gfx1250) — matrix operand order repaired and checked by reading, not compiled, not run on hardware
//
#include <hip/hip_runtime.h>
#include <hip/hip_bf16.h>
#include <hip/amd_detail/amd_hip_unsafe_atomics.h>

// ---------------------------------------------------------------------------
// CDNA5 (gfx1250) implementation.
//  - All dense GEMMs use v_wmma_f32_16x16x32_bf16 (wave32, 16x16 tiles).
//  - Weights pre-packed into WMMA-B fragment order (32B contiguous per lane).
//  - GCN scatter uses global_atomic_add_f32 (unsafeAtomicAdd), 32 lanes/edge.
// ---------------------------------------------------------------------------

typedef __bf16 bf16_t;
typedef bf16_t v16bf __attribute__((ext_vector_type(16)));
typedef bf16_t v8bf  __attribute__((ext_vector_type(8)));
typedef float  v8f   __attribute__((ext_vector_type(8)));

// ---- WMMA helpers ----------------------------------------------------------

__device__ __forceinline__ v8f wmma_bf16(v16bf a, v16bf b, v8f c) {
    // D = A(16x32) * B(32x16) + C  (f32 accum)
    return __builtin_amdgcn_wmma_f32_16x16x32_bf16(
        false, a, false, b, (short)0, c, false, false);
}

// A fragment from a row-major bf16 tile in LDS (16 rows, ldk elements/row).
// ISA layout (16-bit A 16x32): lanes 0-15 hold K {0..7,16..23} of row M=lane,
// lanes 16-31 hold K {8..15,24..31} of row M=lane-16.
__device__ __forceinline__ v16bf load_a_frag(const bf16_t* tile, int ldk,
                                             int kt, int lane) {
    int row   = lane & 15;
    int khalf = (lane & 16) ? 8 : 0;
    const bf16_t* p = tile + row * ldk + kt * 32 + khalf;
    v8bf lo = *(const v8bf*)p;          // K = khalf .. khalf+7
    v8bf hi = *(const v8bf*)(p + 16);   // K = 16+khalf .. 16+khalf+7
    return __builtin_shufflevector(lo, hi, 0,1,2,3,4,5,6,7,8,9,10,11,12,13,14,15);
}

// B fragment from packed weights: tile = kt*nTiles + nt, 512 bf16 per tile,
// lane's 16 values stored contiguously (coalesced 32B per lane).
__device__ __forceinline__ v16bf load_b_frag(const bf16_t* Wp, int tile, int lane) {
    return *(const v16bf*)(Wp + ((size_t)tile << 9) + lane * 16);
}

// ---- weight packing: f32 (Ksrc x Nc) -> bf16 WMMA-B fragment order ---------
// ISA layout (16-bit B 32x16): element i of lane l is W[kt*32+(i&7)+8*(l>=16)+16*(i>=8)][nt*16+(l&15)]
__global__ void pack_w_kernel(const float* __restrict__ W, bf16_t* __restrict__ out,
                              int Ksrc, int Kp, int Nc) {
    int nTile = Nc >> 4;
    long long total = (long long)(Kp >> 5) * nTile * 512;
    long long stride = (long long)gridDim.x * blockDim.x;
    for (long long o = (long long)blockIdx.x * blockDim.x + threadIdx.x;
         o < total; o += stride) {
        int i = (int)(o & 15);
        int l = (int)((o >> 4) & 31);
        long long tile = o >> 9;
        int kt = (int)(tile / nTile), nt = (int)(tile % nTile);
        int k = kt * 32 + (i & 7) + ((l & 16) ? 8 : 0) + ((i & 8) ? 16 : 0);
        int n = nt * 16 + (l & 15);
        float v = (k < Ksrc) ? W[(size_t)k * Nc + n] : 0.0f;
        out[o] = (bf16_t)v;
    }
}

__global__ void zero_kernel(float* __restrict__ p, long long n) {
    long long stride = (long long)gridDim.x * blockDim.x;
    for (long long i = (long long)blockIdx.x * blockDim.x + threadIdx.x;
         i < n; i += stride) p[i] = 0.0f;
}

// ---- Pair branch -----------------------------------------------------------
// One block per batch row b. 4 waves; wave w owns output col-tiles {w, w+4}.
#define PAIR_K 160   // PAIR_IN=131 padded to multiple of 32

__global__ __launch_bounds__(128) void pair_kernel(
    const int* __restrict__ types, const int* __restrict__ lengthes,
    const int* __restrict__ ntypes, const int* __restrict__ ndist,
    const float* __restrict__ nrat, const float* __restrict__ ncom,
    const int* __restrict__ nprice, const float* __restrict__ ngroup,
    const float* __restrict__ Etp, const float* __restrict__ Edp,
    const float* __restrict__ Epp,
    const bf16_t* __restrict__ W1p, const float* __restrict__ b1,
    const bf16_t* __restrict__ W2p, const float* __restrict__ b2,
    const float* __restrict__ Wps1, const float* __restrict__ bps1,
    const float* __restrict__ Wps2, const float* __restrict__ bps2,
    float* __restrict__ pr, int B, int N)
{
    __shared__ __align__(16) bf16_t feat_s[16 * PAIR_K];
    __shared__ __align__(16) bf16_t h1_s[16 * 128];
    __shared__ float te_s[64];
    __shared__ float ctx_s[128];
    __shared__ float ps_s[64];

    const int b    = blockIdx.x;
    const int tid  = threadIdx.x;
    const int lane = tid & 31;
    const int w    = tid >> 5;
    const int len  = lengthes[b];

    if (tid < 64) te_s[tid] = Etp[(size_t)types[b] * 64 + tid];

    const int nloc = lane & 15;
    const int mofs = (lane & 16) ? 8 : 0;
    const int c0 = w * 16 + nloc;
    const int c1 = (w + 4) * 16 + nloc;
    float ctxacc0 = 0.0f, ctxacc1 = 0.0f;

    const int ntiles = (N + 15) >> 4;
    for (int rt = 0; rt < ntiles; ++rt) {
        __syncthreads();  // protect feat_s/h1_s of previous iteration
        // build 16 x PAIR_K bf16 feature tile
        for (int idx = tid; idx < 16 * PAIR_K; idx += 128) {
            int r = idx / PAIR_K, k = idx % PAIR_K;
            int n = rt * 16 + r;
            float v = 0.0f;
            if (n < N) {
                size_t bn = (size_t)b * N + n;
                if (k < 64)        v = Etp[(size_t)ntypes[bn] * 64 + k];
                else if (k < 96)   v = Edp[(size_t)ndist[bn] * 32 + (k - 64)];
                else if (k == 96)  v = nrat[bn];
                else if (k == 97)  v = ncom[bn];
                else if (k < 130)  v = Epp[(size_t)nprice[bn] * 32 + (k - 98)];
                else if (k == 130) v = ngroup[bn];
            }
            feat_s[idx] = (bf16_t)v;
        }
        __syncthreads();
        // stage 1: h1 = relu(feat @ W1 + b1)
        v8f acc0, acc1;
        for (int j = 0; j < 8; ++j) { acc0[j] = 0.0f; acc1[j] = 0.0f; }
        for (int kt = 0; kt < PAIR_K / 32; ++kt) {
            v16bf a  = load_a_frag(feat_s, PAIR_K, kt, lane);
            v16bf bA = load_b_frag(W1p, kt * 8 + w,     lane);
            v16bf bB = load_b_frag(W1p, kt * 8 + w + 4, lane);
            acc0 = wmma_bf16(a, bA, acc0);
            acc1 = wmma_bf16(a, bB, acc1);
        }
        {
            float bb0 = b1[c0], bb1 = b1[c1];
            for (int j = 0; j < 8; ++j) {
                int mr = j + mofs;
                float v0 = acc0[j] + bb0; v0 = v0 > 0.0f ? v0 : 0.0f;
                float v1 = acc1[j] + bb1; v1 = v1 > 0.0f ? v1 : 0.0f;
                h1_s[mr * 128 + c0] = (bf16_t)v0;
                h1_s[mr * 128 + c1] = (bf16_t)v1;
            }
        }
        __syncthreads();
        // stage 2: h2 = relu(h1 @ W2 + b2), masked accumulate into context
        for (int j = 0; j < 8; ++j) { acc0[j] = 0.0f; acc1[j] = 0.0f; }
        for (int kt = 0; kt < 4; ++kt) {
            v16bf a  = load_a_frag(h1_s, 128, kt, lane);
            v16bf bA = load_b_frag(W2p, kt * 8 + w,     lane);
            v16bf bB = load_b_frag(W2p, kt * 8 + w + 4, lane);
            acc0 = wmma_bf16(a, bA, acc0);
            acc1 = wmma_bf16(a, bB, acc1);
        }
        {
            float bb0 = b2[c0], bb1 = b2[c1];
            float p0 = 0.0f, p1 = 0.0f;
            for (int j = 0; j < 8; ++j) {
                int n = rt * 16 + j + mofs;
                float m = (n < len) ? 1.0f : 0.0f;
                float v0 = acc0[j] + bb0; v0 = v0 > 0.0f ? v0 : 0.0f;
                float v1 = acc1[j] + bb1; v1 = v1 > 0.0f ? v1 : 0.0f;
                p0 += m * v0; p1 += m * v1;
            }
            p0 += __shfl_down(p0, 16, 32);
            p1 += __shfl_down(p1, 16, 32);
            if (lane < 16) { ctxacc0 += p0; ctxacc1 += p1; }
        }
    }
    if (lane < 16) {
        float denom = (float)(len > 1 ? len : 1);
        ctx_s[w * 16 + lane]       = ctxacc0 / denom;
        ctx_s[(w + 4) * 16 + lane] = ctxacc1 / denom;
    }
    __syncthreads();
    // ps head: ps = relu([ctx, te] @ Wps1 + bps1); pr = ps @ Wps2 + bps2
    if (tid < 64) {
        float acc = bps1[tid];
        for (int k = 0; k < 128; ++k) acc += ctx_s[k] * Wps1[(size_t)k * 64 + tid];
        for (int k = 0; k < 64;  ++k) acc += te_s[k]  * Wps1[(size_t)(128 + k) * 64 + tid];
        acc = acc > 0.0f ? acc : 0.0f;
        ps_s[tid] = acc * Wps2[tid];
    }
    __syncthreads();
    if (tid == 0) {
        float s = bps2[0];
        for (int k = 0; k < 64; ++k) s += ps_s[k];
        pr[b] = s;
    }
}

// ---- GNN: node feature GEMM x = relu(xg @ W_gs + b_gs) ---------------------
__global__ __launch_bounds__(128) void node_feat_kernel(
    const int* __restrict__ all_types, const int* __restrict__ all_prices,
    const float* __restrict__ all_groups,
    const float* __restrict__ Etg, const float* __restrict__ Epg,
    const bf16_t* __restrict__ Wgsp, const float* __restrict__ bgs,
    bf16_t* __restrict__ X, int M)
{
    __shared__ __align__(16) bf16_t a_s[16 * 128];
    const int m0 = blockIdx.x * 16;
    const int tid = threadIdx.x, lane = tid & 31, w = tid >> 5;
    for (int idx = tid; idx < 16 * 128; idx += 128) {
        int r = idx >> 7, k = idx & 127;
        int m = m0 + r;
        float v = 0.0f;
        if (m < M) {
            if (k < 64)       v = Etg[(size_t)all_types[m] * 64 + k];
            else if (k < 96)  v = Epg[(size_t)all_prices[m] * 32 + (k - 64)];
            else if (k == 96) v = all_groups[m];
        }
        a_s[idx] = (bf16_t)v;
    }
    __syncthreads();
    const int nloc = lane & 15, mofs = (lane & 16) ? 8 : 0;
    for (int t = 0; t < 2; ++t) {
        int nt = w + t * 4;
        v8f acc; for (int j = 0; j < 8; ++j) acc[j] = 0.0f;
        for (int kt = 0; kt < 4; ++kt) {
            v16bf a  = load_a_frag(a_s, 128, kt, lane);
            v16bf bb = load_b_frag(Wgsp, kt * 8 + nt, lane);
            acc = wmma_bf16(a, bb, acc);
        }
        int col = nt * 16 + nloc;
        float bias = bgs[col];
        for (int j = 0; j < 8; ++j) {
            int m = m0 + j + mofs;
            if (m < M) {
                float v = acc[j] + bias; v = v > 0.0f ? v : 0.0f;
                X[(size_t)m * 128 + col] = (bf16_t)v;
            }
        }
    }
}

// ---- generic 128-K GEMM: T = Z(bf16, Mx128) @ Wp(packed 128x128) -> f32 ----
__global__ __launch_bounds__(128) void gemm128_kernel(
    const bf16_t* __restrict__ Z, const bf16_t* __restrict__ Wp,
    float* __restrict__ T, int M)
{
    __shared__ __align__(16) bf16_t z_s[16 * 128];
    const int m0 = blockIdx.x * 16;
    const int tid = threadIdx.x, lane = tid & 31, w = tid >> 5;
    for (int idx = tid; idx < 16 * 128 / 8; idx += 128) {
        int r = idx >> 4;
        int kc = (idx & 15) * 8;
        int m = m0 + r;
        v8bf v;
        if (m < M) v = *(const v8bf*)(Z + (size_t)m * 128 + kc);
        else for (int j = 0; j < 8; ++j) v[j] = (bf16_t)0.0f;
        *(v8bf*)(z_s + r * 128 + kc) = v;
    }
    __syncthreads();
    const int nloc = lane & 15, mofs = (lane & 16) ? 8 : 0;
    for (int t = 0; t < 2; ++t) {
        int nt = w + t * 4;
        v8f acc; for (int j = 0; j < 8; ++j) acc[j] = 0.0f;
        for (int kt = 0; kt < 4; ++kt) {
            v16bf a  = load_a_frag(z_s, 128, kt, lane);
            v16bf bb = load_b_frag(Wp, kt * 8 + nt, lane);
            acc = wmma_bf16(a, bb, acc);
        }
        int col = nt * 16 + nloc;
        for (int j = 0; j < 8; ++j) {
            int m = m0 + j + mofs;
            if (m < M) T[(size_t)m * 128 + col] = acc[j];
        }
    }
}

// ---- degree / dinv ---------------------------------------------------------
__global__ void deg_kernel(const int* __restrict__ edge_index,
                           float* __restrict__ deg, int E) {
    int e = blockIdx.x * blockDim.x + threadIdx.x;
    if (e < E) unsafeAtomicAdd(&deg[edge_index[(size_t)E + e]], 1.0f);
}

__global__ void dinv_kernel(float* __restrict__ deg, int M) {
    int i = blockIdx.x * blockDim.x + threadIdx.x;
    if (i < M) deg[i] = rsqrtf(deg[i] + 1.0f);  // +1 self loop; always >= 1
}

// ---- edge scatter: Y[dst] += T[src] * dinv[src]*dinv[dst] ------------------
// 32 lanes per edge, 4 cols per lane (float4 gather + 4 f32 global atomics)
__global__ void scatter_kernel(const float* __restrict__ T,
                               const int* __restrict__ edge_index,
                               const float* __restrict__ dinv,
                               float* __restrict__ Y, int E, int M,
                               const int* __restrict__ bidp, int Bsz, int windowed)
{
    long long g = (long long)blockIdx.x * blockDim.x + threadIdx.x;
    long long edge = g >> 5;
    int lane = (int)(g & 31);
    long long total = (long long)E + M;
    if (edge >= total) return;
    int s, d;
    if (edge < E) { s = edge_index[edge]; d = edge_index[(long long)E + edge]; }
    else          { s = d = (int)(edge - E); }
    if (windowed) {
        int lo = bidp[0] * Bsz;
        if (d < lo || d >= lo + Bsz) return;
    }
    float norm = dinv[s] * dinv[d];
    const float4 t4 = *(const float4*)(T + (size_t)s * 128 + lane * 4);
    float* yp = Y + (size_t)d * 128 + lane * 4;
    unsafeAtomicAdd(yp + 0, t4.x * norm);
    unsafeAtomicAdd(yp + 1, t4.y * norm);
    unsafeAtomicAdd(yp + 2, t4.z * norm);
    unsafeAtomicAdd(yp + 3, t4.w * norm);
}

// ---- g1 = relu(Y + b) -> bf16 ----------------------------------------------
__global__ void bias_act_bf16_kernel(const float* __restrict__ Y,
                                     const float* __restrict__ bias,
                                     bf16_t* __restrict__ Out, long long total)
{
    long long stride = (long long)gridDim.x * blockDim.x;
    for (long long i = (long long)blockIdx.x * blockDim.x + threadIdx.x;
         i < total; i += stride) {
        int col = (int)(i & 127);
        float v = Y[i] + bias[col];
        v = v > 0.0f ? v : 0.0f;
        Out[i] = (bf16_t)v;
    }
}

// ---- head: q=relu((Y2+b_gc2)[window] @ Wq1 + bq1); out = combine ----------
__global__ __launch_bounds__(128) void head_kernel(
    const float* __restrict__ Y2, const float* __restrict__ bgc2,
    const bf16_t* __restrict__ Wq1p, const float* __restrict__ bq1,
    const float* __restrict__ Wq2, const float* __restrict__ bq2,
    const float* __restrict__ Wc, const float* __restrict__ bc,
    const float* __restrict__ pr, const int* __restrict__ bidp,
    float* __restrict__ out, int B, int M)
{
    __shared__ __align__(16) bf16_t a_s[16 * 128];
    __shared__ float q_s[16 * 64];
    const int r0 = blockIdx.x * 16;
    const int lo = bidp[0] * B;
    const int tid = threadIdx.x, lane = tid & 31, w = tid >> 5;
    for (int idx = tid; idx < 16 * 128; idx += 128) {
        int r = idx >> 7, k = idx & 127;
        int row = r0 + r;
        int gidx = lo + row;
        float v = 0.0f;
        if (row < B && gidx < M) v = Y2[(size_t)gidx * 128 + k] + bgc2[k];
        a_s[idx] = (bf16_t)v;
    }
    __syncthreads();
    {
        int nt = w;  // 4 col tiles of 16 -> 64 cols
        v8f acc; for (int j = 0; j < 8; ++j) acc[j] = 0.0f;
        for (int kt = 0; kt < 4; ++kt) {
            v16bf a  = load_a_frag(a_s, 128, kt, lane);
            v16bf bb = load_b_frag(Wq1p, kt * 4 + nt, lane);
            acc = wmma_bf16(a, bb, acc);
        }
        int nloc = lane & 15, mofs = (lane & 16) ? 8 : 0;
        int col = nt * 16 + nloc;
        float bias = bq1[col];
        for (int j = 0; j < 8; ++j) {
            float v = acc[j] + bias; v = v > 0.0f ? v : 0.0f;
            q_s[(j + mofs) * 64 + col] = v;
        }
    }
    __syncthreads();
    if (tid < 16) {
        int row = r0 + tid;
        if (row < B) {
            float s = bq2[0];
            for (int k = 0; k < 64; ++k) s += q_s[tid * 64 + k] * Wq2[k];
            out[row] = pr[row] * Wc[0] + s * Wc[1] + bc[0];
        }
    }
}

// ---------------------------------------------------------------------------
extern "C" void kernel_launch(void* const* d_in, const int* in_sizes, int n_in,
                              void* d_out, int out_size, void* d_ws, size_t ws_size,
                              hipStream_t stream) {
    const int*   types    = (const int*)d_in[0];
    const int*   lengthes = (const int*)d_in[4];
    const int*   ntypes   = (const int*)d_in[5];
    const int*   ndist    = (const int*)d_in[6];
    const float* nrat     = (const float*)d_in[7];
    const float* ncom     = (const float*)d_in[8];
    const int*   nprice   = (const int*)d_in[9];
    const float* ngroup   = (const float*)d_in[10];
    const int*   all_types  = (const int*)d_in[11];
    const int*   all_prices = (const int*)d_in[12];
    const float* all_groups = (const float*)d_in[13];
    const int*   edge_index = (const int*)d_in[14];
    const int*   bidp       = (const int*)d_in[15];
    const float* Etp  = (const float*)d_in[16];
    const float* Edp  = (const float*)d_in[17];
    const float* Epp  = (const float*)d_in[18];
    const float* Wsh1 = (const float*)d_in[19];
    const float* bsh1 = (const float*)d_in[20];
    const float* Wsh2 = (const float*)d_in[21];
    const float* bsh2 = (const float*)d_in[22];
    const float* Wps1 = (const float*)d_in[23];
    const float* bps1 = (const float*)d_in[24];
    const float* Wps2 = (const float*)d_in[25];
    const float* bps2 = (const float*)d_in[26];
    const float* Etg  = (const float*)d_in[27];
    const float* Epg  = (const float*)d_in[28];
    const float* Wgs  = (const float*)d_in[29];
    const float* bgs  = (const float*)d_in[30];
    const float* Wgc1 = (const float*)d_in[31];
    const float* bgc1 = (const float*)d_in[32];
    const float* Wgc2 = (const float*)d_in[33];
    const float* bgc2 = (const float*)d_in[34];
    const float* Wq1  = (const float*)d_in[35];
    const float* bq1  = (const float*)d_in[36];
    const float* Wq2  = (const float*)d_in[37];
    const float* bq2  = (const float*)d_in[38];
    const float* Wc   = (const float*)d_in[39];
    const float* bc   = (const float*)d_in[40];
    float* out = (float*)d_out;

    const int B = in_sizes[0];
    const int N = in_sizes[5] / B;
    const int M = in_sizes[11];
    const int E = in_sizes[14] / 2;

    // workspace carve (256B aligned)
    char* p = (char*)d_ws;
    auto carve = [&](size_t bytes) -> void* {
        void* r = (void*)p;
        p += (bytes + 255) & ~(size_t)255;
        return r;
    };
    auto packedBytes = [](int Kp, int Nc) -> size_t {
        return (size_t)(Kp / 32) * (Nc / 16) * 512 * sizeof(bf16_t);
    };
    bf16_t* w1p   = (bf16_t*)carve(packedBytes(PAIR_K, 128));
    bf16_t* w2p   = (bf16_t*)carve(packedBytes(128, 128));
    bf16_t* wgsp  = (bf16_t*)carve(packedBytes(128, 128));
    bf16_t* wgc1p = (bf16_t*)carve(packedBytes(128, 128));
    bf16_t* wgc2p = (bf16_t*)carve(packedBytes(128, 128));
    bf16_t* wq1p  = (bf16_t*)carve(packedBytes(128, 64));
    float*  pr    = (float*)carve((size_t)B * sizeof(float));
    float*  dinv  = (float*)carve((size_t)M * sizeof(float));
    bf16_t* X     = (bf16_t*)carve((size_t)M * 128 * sizeof(bf16_t));
    bf16_t* G1    = (bf16_t*)carve((size_t)M * 128 * sizeof(bf16_t));
    float*  T     = (float*)carve((size_t)M * 128 * sizeof(float));
    float*  Y     = (float*)carve((size_t)M * 128 * sizeof(float));

    // 1) pack weights into WMMA-B fragment layout (bf16)
    pack_w_kernel<<<160, 256, 0, stream>>>(Wsh1, w1p, 131, PAIR_K, 128);
    pack_w_kernel<<<128, 256, 0, stream>>>(Wsh2, w2p, 128, 128, 128);
    pack_w_kernel<<<128, 256, 0, stream>>>(Wgs,  wgsp, 97, 128, 128);
    pack_w_kernel<<<128, 256, 0, stream>>>(Wgc1, wgc1p, 128, 128, 128);
    pack_w_kernel<<<128, 256, 0, stream>>>(Wgc2, wgc2p, 128, 128, 128);
    pack_w_kernel<<<64,  256, 0, stream>>>(Wq1,  wq1p, 128, 128, 64);

    // 2) pair branch
    pair_kernel<<<B, 128, 0, stream>>>(types, lengthes, ntypes, ndist, nrat, ncom,
                                       nprice, ngroup, Etp, Edp, Epp,
                                       w1p, bsh1, w2p, bsh2,
                                       Wps1, bps1, Wps2, bps2, pr, B, N);

    // 3) GNN branch
    const int mTiles = (M + 15) / 16;
    node_feat_kernel<<<mTiles, 128, 0, stream>>>(all_types, all_prices, all_groups,
                                                 Etg, Epg, wgsp, bgs, X, M);

    zero_kernel<<<1024, 256, 0, stream>>>(dinv, (long long)M);
    deg_kernel<<<(E + 255) / 256, 256, 0, stream>>>(edge_index, dinv, E);
    dinv_kernel<<<(M + 255) / 256, 256, 0, stream>>>(dinv, M);

    const long long totEdges = (long long)E + M;
    const int scatterBlocks = (int)((totEdges * 32 + 255) / 256);

    // layer 1
    gemm128_kernel<<<mTiles, 128, 0, stream>>>(X, wgc1p, T, M);
    zero_kernel<<<4096, 256, 0, stream>>>(Y, (long long)M * 128);
    scatter_kernel<<<scatterBlocks, 256, 0, stream>>>(T, edge_index, dinv, Y, E, M,
                                                      bidp, B, 0);
    bias_act_bf16_kernel<<<4096, 256, 0, stream>>>(Y, bgc1, G1, (long long)M * 128);

    // layer 2 (only the batch window of dst rows feeds the head)
    gemm128_kernel<<<mTiles, 128, 0, stream>>>(G1, wgc2p, T, M);
    zero_kernel<<<4096, 256, 0, stream>>>(Y, (long long)M * 128);
    scatter_kernel<<<scatterBlocks, 256, 0, stream>>>(T, edge_index, dinv, Y, E, M,
                                                      bidp, B, 1);

    // 4) head + combine -> d_out
    head_kernel<<<(B + 15) / 16, 128, 0, stream>>>(Y, bgc2, wq1p, bq1, Wq2, bq2,
                                                   Wc, bc, pr, bidp, out, B, M);
}
